// GravnetModel_49813030699586
// MI455X (gfx1250) — compile-verified, matrix operand order checked by reading, not run on hardware
//
#include <hip/hip_runtime.h>
#include <hip/hip_bf16.h>
#include <cstdint>

// ---------------- model constants ----------------
#define NHITS   32768   // total hits
#define NPE     2048    // hits per event
#define NEV     16      // events
#define KNN     40      // neighbours (includes self)
#define INF_F   3.4e38f

typedef __attribute__((ext_vector_type(16))) _Float16 v16h;
typedef __attribute__((ext_vector_type(8)))  _Float16 v8h;
typedef __attribute__((ext_vector_type(4)))  _Float16 v4h;
typedef __attribute__((ext_vector_type(8)))  float    v8f;

// K-index mapping for 16-bit A/B WMMA fragments (ISA 7.12.2, 16x32):
//   lanes 0-15 : elem 0..7 -> K=0..7,   elem 8..15 -> K=16..23
//   lanes16-31 : elem 0..7 -> K=8..15,  elem 8..15 -> K=24..31
__device__ __host__ __forceinline__ int kmap(int i, int lane) {
    int base = (lane < 16) ? 0 : 8;
    return (i < 8) ? (base + i) : (base + 16 + (i - 8));
}

// =======================================================================
// Weight pre-pack: W (K x Nout f32, row-major) -> fragment-ordered f16.
// Layout: [tile t][K-step kki][lane 0..31][elem 0..15], zero-padded.
// A wave then loads its whole B fragment as 2 coalesced 16B loads/lane.
// =======================================================================
__global__ void pack_w_kernel(const float* __restrict__ W, _Float16* __restrict__ out,
                              int K, int Nout)
{
    const int g = (int)blockIdx.x * 256 + threadIdx.x;
    const int nK = (K + 31) >> 5;
    const int nT = (Nout + 15) >> 4;
    const int total = nT * nK * 512;
    if (g >= total) return;
    const int e    = g & 15;
    const int lane = (g >> 4) & 31;
    const int kki  = (g >> 9) % nK;
    const int t    = g / (nK * 512);
    const int k = (kki << 5) + kmap(e, lane);
    const int n = t * 16 + (lane & 15);
    _Float16 v = (_Float16)0.f;
    if (k < K && n < Nout) v = (_Float16)W[(size_t)k * Nout + n];
    out[g] = v;
}

// =======================================================================
// Fused GEMM:  C = epilogue( concat(A0,A1) x W + bias )
//   Block: 256 thr (8 waves), 32 rows (2 M-tiles); wave w -> N-tile w.
//   Per K-step: cooperative LDS staging of 32x32 f16 A tile (preBN fused),
//   then per wave: 4x ds_load_b128 (A frags) + 2x 16B global (B frag),
//   2x v_wmma_f32_16x16x32_f16.
// Template: ACT 0 none/1 tanh/2 relu; PRE_BN; TWO_SEG (A1 covers [k0,K));
//   EVT (A0 row -> row>>11); POST_BN; HAS_C1 (dual output).
// =======================================================================
template<int ACT, bool PRE_BN, bool TWO_SEG, bool EVT, bool POST_BN, bool HAS_C1>
__global__ __launch_bounds__(256)
void gemm_wmma_kernel(const float* __restrict__ A0, int lda0, int k0,
                      const float* __restrict__ A1, int lda1,
                      const float* __restrict__ preBN,
                      const _Float16* __restrict__ Wpk,
                      const float* __restrict__ bias,
                      int K, int Nout,
                      const float* __restrict__ postBN,
                      float* __restrict__ C0, int ldc0, int c0off,
                      float* __restrict__ C1, int ldc1)
{
    __shared__ __align__(16) _Float16 atile[32 * 32];   // 2 M-tiles x 32 K (2KB)
    const int tid  = threadIdx.x;
    const int wave = tid >> 5;
    const int lane = tid & 31;
    const int m0   = (int)blockIdx.x * 32;
    const int nK   = (K + 31) >> 5;
    const int n0   = wave * 16;
    const bool active = (n0 < Nout);              // wave-uniform

    // staging coords: each thread stages 4 consecutive halves of one row
    const int srow = tid >> 3;                    // 0..31
    const int sk0  = (tid & 7) * 4;               // 0,4,..,28
    const int grow = m0 + srow;

    // fragment read coords
    const int r0   = lane & 15;
    const int kb   = (lane < 16) ? 0 : 8;
    const int col  = n0 + r0;

    v8f acc0 = {0.f,0.f,0.f,0.f,0.f,0.f,0.f,0.f};
    v8f acc1 = acc0;

    for (int kki = 0; kki < nK; ++kki) {
        const int kbase = kki << 5;
        // ---- gather + convert 4 halves (runs on all 256 threads) ----
        v4h h4;
#pragma unroll
        for (int j = 0; j < 4; ++j) {
            const int k = kbase + sk0 + j;
            float av = 0.f;
            if (k < K) {
                if (TWO_SEG && k >= k0) {
                    av = A1[(size_t)grow * lda1 + (k - k0)];
                } else {
                    const int r = EVT ? (grow >> 11) : grow;
                    av = A0[(size_t)r * lda0 + k];
                }
                if (PRE_BN) av = av * preBN[2 * k] + preBN[2 * k + 1];
            }
            h4[j] = (_Float16)av;
        }
        __syncthreads();                          // previous tile fully consumed
        *(v4h*)(atile + srow * 32 + sk0) = h4;    // ds_store_b64
        __syncthreads();                          // tile ready

        if (active) {
            // A fragments from LDS: 16B ds loads, layout matches kmap
            v8h a0lo = *(const v8h*)(atile + r0 * 32 + kb);
            v8h a0hi = *(const v8h*)(atile + r0 * 32 + kb + 16);
            v8h a1lo = *(const v8h*)(atile + (16 + r0) * 32 + kb);
            v8h a1hi = *(const v8h*)(atile + (16 + r0) * 32 + kb + 16);
            // B fragment: coalesced 32B per lane from packed arena
            const _Float16* bp = Wpk + (((size_t)wave * nK + kki) * 32 + lane) * 16;
            v8h blo = *(const v8h*)bp;
            v8h bhi = *(const v8h*)(bp + 8);

            v16h afrag, bfrag;
#pragma unroll
            for (int i = 0; i < 8; ++i) {
                afrag[i] = a0lo[i]; afrag[8 + i] = a0hi[i];
                bfrag[i] = blo[i];  bfrag[8 + i] = bhi[i];
            }
            acc0 = __builtin_amdgcn_wmma_f32_16x16x32_f16(
                       false, afrag, false, bfrag, (short)0, acc0, false, false);
#pragma unroll
            for (int i = 0; i < 8; ++i) {
                afrag[i] = a1lo[i]; afrag[8 + i] = a1hi[i];
            }
            acc1 = __builtin_amdgcn_wmma_f32_16x16x32_f16(
                       false, afrag, false, bfrag, (short)0, acc1, false, false);
        }
    }

    if (active && col < Nout) {
        const float bs = bias[col];
        const float ps = POST_BN ? postBN[2 * col]     : 1.f;
        const float ph = POST_BN ? postBN[2 * col + 1] : 0.f;
#pragma unroll
        for (int r = 0; r < 8; ++r) {
            const int row = m0 + r + ((lane < 16) ? 0 : 8);
#pragma unroll
            for (int half = 0; half < 2; ++half) {
                float y = (half ? acc1[r] : acc0[r]) + bs;
                if (ACT == 1)      y = tanhf(y);
                else if (ACT == 2) y = fmaxf(y, 0.f);
                if (POST_BN)       y = y * ps + ph;
                const int rr = row + half * 16;
                C0[(size_t)rr * ldc0 + c0off + col] = y;
                if (HAS_C1) C1[(size_t)rr * ldc1 + col] = y;
            }
        }
    }
}

// =======================================================================
// BN fold:  sc = g * rsqrt(v+eps); sh = b - m*sc   -> out[2i],out[2i+1]
// =======================================================================
__global__ void bn_prep_kernel(const float* __restrict__ g, const float* __restrict__ b,
                               const float* __restrict__ m, const float* __restrict__ v,
                               float* __restrict__ out, int d)
{
    int i = threadIdx.x;
    if (i < d) {
        float sc = g[i] * rsqrtf(v[i] + 1e-5f);
        out[2 * i]     = sc;
        out[2 * i + 1] = b[i] - m[i] * sc;
    }
}

// =======================================================================
// Input stage: per-event mean/min/max of bn1(x) (5 dims) -> stats5[e][15]
// =======================================================================
__global__ __launch_bounds__(256)
void stats5_kernel(const float* __restrict__ x, const float* __restrict__ bn1,
                   float* __restrict__ stats5)
{
    __shared__ float buf[256][15];
    const int e = blockIdx.x, t = threadIdx.x;
    float sm[5], mn[5], mx[5];
#pragma unroll
    for (int d = 0; d < 5; ++d) { sm[d] = 0.f; mn[d] = INF_F; mx[d] = -INF_F; }
    for (int i = t; i < NPE; i += 256) {
#pragma unroll
        for (int d = 0; d < 5; ++d) {
            float v = x[(size_t)(e * NPE + i) * 5 + d] * bn1[2 * d] + bn1[2 * d + 1];
            sm[d] += v; mn[d] = fminf(mn[d], v); mx[d] = fmaxf(mx[d], v);
        }
    }
#pragma unroll
    for (int d = 0; d < 5; ++d) { buf[t][d] = sm[d]; buf[t][5 + d] = mn[d]; buf[t][10 + d] = mx[d]; }
    __syncthreads();
    for (int s = 128; s > 0; s >>= 1) {
        if (t < s) {
#pragma unroll
            for (int d = 0; d < 5; ++d) {
                buf[t][d]      += buf[t + s][d];
                buf[t][5 + d]   = fminf(buf[t][5 + d],  buf[t + s][5 + d]);
                buf[t][10 + d]  = fmaxf(buf[t][10 + d], buf[t + s][10 + d]);
            }
        }
        __syncthreads();
    }
    if (t < 15) stats5[e * 15 + t] = (t < 5) ? buf[0][t] / (float)NPE : buf[0][t];
}

// input linear: [mean5,min5,max5,bn1(x)5] (20) -> 64
__global__ void inp_kernel(const float* __restrict__ x, const float* __restrict__ bn1,
                           const float* __restrict__ stats5,
                           const float* __restrict__ W, const float* __restrict__ bi,
                           float* __restrict__ out)
{
    int i = (int)blockIdx.x * 256 + threadIdx.x;
    if (i >= NHITS) return;
    const int e = i >> 11;
    float v[20];
#pragma unroll
    for (int d = 0; d < 15; ++d) v[d] = stats5[e * 15 + d];
#pragma unroll
    for (int d = 0; d < 5; ++d)
        v[15 + d] = x[(size_t)i * 5 + d] * bn1[2 * d] + bn1[2 * d + 1];
    for (int o = 0; o < 64; ++o) {
        float acc = bi[o];
#pragma unroll
        for (int k = 0; k < 20; ++k) acc += v[k] * W[k * 64 + o];
        out[(size_t)i * 64 + o] = acc;
    }
}

// =======================================================================
// lin_s (cin->4) and lin_h (cin->22) in one pass
// =======================================================================
__global__ void lin_sh_kernel(const float* __restrict__ x, int cin,
                              const float* __restrict__ Ws, const float* __restrict__ bs,
                              const float* __restrict__ Wh, const float* __restrict__ bh,
                              float* __restrict__ s, float* __restrict__ h)
{
    int i = (int)blockIdx.x * 256 + threadIdx.x;
    if (i >= NHITS) return;
    float acc[26];
#pragma unroll
    for (int o = 0; o < 4; ++o)  acc[o]     = bs[o];
#pragma unroll
    for (int o = 0; o < 22; ++o) acc[4 + o] = bh[o];
    for (int k = 0; k < cin; ++k) {
        float xv = x[(size_t)i * cin + k];
#pragma unroll
        for (int o = 0; o < 4; ++o)  acc[o]     += xv * Ws[k * 4 + o];
#pragma unroll
        for (int o = 0; o < 22; ++o) acc[4 + o] += xv * Wh[k * 22 + o];
    }
#pragma unroll
    for (int o = 0; o < 4; ++o)  s[(size_t)i * 4 + o]  = acc[o];
#pragma unroll
    for (int o = 0; o < 22; ++o) h[(size_t)i * 22 + o] = acc[4 + o];
}

// =======================================================================
// kNN: one wave per query row. Distance row (2048 f32) in LDS, then 40
// iterations of wave-parallel argmin with masking. 4 waves / block.
// =======================================================================
__global__ __launch_bounds__(128)
void knn_kernel(const float* __restrict__ s, int* __restrict__ idx,
                float* __restrict__ wgt)
{
    __shared__ float dist[4][NPE];
    const int wave = threadIdx.x >> 5, lane = threadIdx.x & 31;
    const int e  = (int)blockIdx.x >> 9;               // 512 blocks / event
    const int r  = (((int)blockIdx.x & 511) << 2) + wave;
    const int gr = e * NPE + r;

    const float4  sr = ((const float4*)s)[gr];
    const float4* se = (const float4*)s + (size_t)e * NPE;
    for (int c = lane; c < NPE; c += 32) {
        float4 sc = se[c];
        float dx = sr.x - sc.x, dy = sr.y - sc.y, dz = sr.z - sc.z, dw = sr.w - sc.w;
        dist[wave][c] = dx * dx + dy * dy + dz * dz + dw * dw;
    }
    __syncthreads();

    for (int t = 0; t < KNN; ++t) {
        float best = INF_F; int bi = 0;
        for (int c = lane; c < NPE; c += 32) {
            float d = dist[wave][c];
            if (d < best) { best = d; bi = c; }
        }
#pragma unroll
        for (int off = 16; off > 0; off >>= 1) {
            float ob = __shfl_xor(best, off, 32);
            int   oi = __shfl_xor(bi,   off, 32);
            if (ob < best || (ob == best && oi < bi)) { best = ob; bi = oi; }
        }
        if (lane == 0) {
            idx[(size_t)gr * KNN + t] = e * NPE + bi;
            wgt[(size_t)gr * KNN + t] = __expf(-10.0f * best);
            dist[wave][bi] = INF_F;
        }
        __syncthreads();
    }
}

// =======================================================================
// message aggregation: agg = [mean_k(h[idx]*w), max_k(h[idx]*w)]  (44)
// =======================================================================
__global__ void aggregate_kernel(const float* __restrict__ h, const int* __restrict__ idx,
                                 const float* __restrict__ wgt, float* __restrict__ agg)
{
    int i = (int)blockIdx.x * 256 + threadIdx.x;
    if (i >= NHITS) return;
    float sm[22], mx[22];
#pragma unroll
    for (int d = 0; d < 22; ++d) { sm[d] = 0.f; mx[d] = -INF_F; }
    for (int k = 0; k < KNN; ++k) {
        const int   j = idx[(size_t)i * KNN + k];
        const float w = wgt[(size_t)i * KNN + k];
#pragma unroll
        for (int d = 0; d < 22; ++d) {
            float m = h[(size_t)j * 22 + d] * w;
            sm[d] += m; mx[d] = fmaxf(mx[d], m);
        }
    }
#pragma unroll
    for (int d = 0; d < 22; ++d) {
        agg[(size_t)i * 44 + d]      = sm[d] * (1.0f / KNN);
        agg[(size_t)i * 44 + 22 + d] = mx[d];
    }
}

// =======================================================================
// per-event mean/min/max of a 96-dim tensor -> stats[e][288]
// =======================================================================
__global__ __launch_bounds__(256)
void stats96_kernel(const float* __restrict__ x, float* __restrict__ stats)
{
    __shared__ float bs[256], bn_[256], bx[256];
    const int e = (int)blockIdx.x / 96, d = (int)blockIdx.x % 96;
    const int t = threadIdx.x;
    float sm = 0.f, mn = INF_F, mx = -INF_F;
    for (int i = t; i < NPE; i += 256) {
        float v = x[(size_t)(e * NPE + i) * 96 + d];
        sm += v; mn = fminf(mn, v); mx = fmaxf(mx, v);
    }
    bs[t] = sm; bn_[t] = mn; bx[t] = mx;
    __syncthreads();
    for (int s = 128; s > 0; s >>= 1) {
        if (t < s) {
            bs[t] += bs[t + s];
            bn_[t] = fminf(bn_[t], bn_[t + s]);
            bx[t]  = fmaxf(bx[t],  bx[t + s]);
        }
        __syncthreads();
    }
    if (t == 0) {
        stats[e * 288 + d]       = bs[0] / (float)NPE;
        stats[e * 288 + 96 + d]  = bn_[0];
        stats[e * 288 + 192 + d] = bx[0];
    }
}

// =======================================================================
// host orchestration
// =======================================================================
#define GEMM_GRID  (NHITS / 32)

extern "C" void kernel_launch(void* const* d_in, const int* in_sizes, int n_in,
                              void* d_out, int out_size, void* d_ws, size_t ws_size,
                              hipStream_t stream)
{
    // ---- inputs (setup_inputs dict order): x, batch, then params leaves ----
    const float* x_in = (const float*)d_in[0];
    (void)d_in[1];  // batch: structurally known (i >> 11)
    int pi = 2;
    auto nxt = [&]() { return (const float*)d_in[pi++]; };

    struct Lin { const float* w; const float* b; };
    struct Bn  { const float* g; const float* b; const float* m; const float* v; };

    Bn  bn1 { nxt(), nxt(), nxt(), nxt() };
    Lin inp { nxt(), nxt() };
    struct BlockP { Lin ls, lh, lo; Bn ba; Lin fa; Bn bb; Lin fb; Lin fo; Bn bo; } BP[4];
    for (int b = 0; b < 4; ++b) {
        BP[b].ls = { nxt(), nxt() };
        BP[b].lh = { nxt(), nxt() };
        BP[b].lo = { nxt(), nxt() };
        BP[b].ba = { nxt(), nxt(), nxt(), nxt() };
        BP[b].fa = { nxt(), nxt() };
        BP[b].bb = { nxt(), nxt(), nxt(), nxt() };
        BP[b].fb = { nxt(), nxt() };
        BP[b].fo = { nxt(), nxt() };
        BP[b].bo = { nxt(), nxt(), nxt(), nxt() };
    }
    struct PG { Lin fc; Bn bn; } PGs[4];
    for (int i = 0; i < 4; ++i)
        PGs[i] = { { nxt(), nxt() }, { nxt(), nxt(), nxt(), nxt() } };
    Lin o1 { nxt(), nxt() }, o2 { nxt(), nxt() }, o3 { nxt(), nxt() };

    // ---- workspace carve ----
    uintptr_t wp = (uintptr_t)d_ws;
    auto carve = [&](size_t bytes) -> void* {
        void* p = (void*)wp; wp += (bytes + 255) & ~(size_t)255; return p;
    };
    const size_t NHs = NHITS;
    float* cur   = (float*)carve(NHs * 96  * 4);   // block input (64 cols stage0)
    float* x96   = (float*)carve(NHs * 96  * 4);   // conv out / fc_b out
    float* t128a = (float*)carve(NHs * 128 * 4);
    float* t128b = (float*)carve(NHs * 128 * 4);
    float* outs  = (float*)carve(NHs * 384 * 4);   // concat of block outputs
    float* s4    = (float*)carve(NHs * 4   * 4);
    float* h22   = (float*)carve(NHs * 22  * 4);
    float* agg   = (float*)carve(NHs * 44  * 4);
    float* wgt   = (float*)carve(NHs * KNN * 4);
    int*   idx   = (int*)  carve(NHs * KNN * 4);
    float* t64a  = (float*)carve(NHs * 64  * 4);
    float* t64b  = (float*)carve(NHs * 64  * 4);
    float* st5   = (float*)carve(NEV * 15  * 4);
    float* st96  = (float*)carve(NEV * 288 * 4);
    float* bnp1  = (float*)carve(5   * 2 * 4);
    float* bnpA[4], *bnpB[4], *bnpO[4], *bnpP[4];
    for (int b = 0; b < 4; ++b) {
        bnpA[b] = (float*)carve(96  * 2 * 4);
        bnpB[b] = (float*)carve(128 * 2 * 4);
        bnpO[b] = (float*)carve(96  * 2 * 4);
    }
    for (int i = 0; i < 4; ++i) bnpP[i] = (float*)carve(128 * 2 * 4);
    _Float16* arena = (_Float16*)carve(2u << 20);  // packed f16 weight arena
    (void)ws_size; (void)in_sizes; (void)n_in; (void)out_size;

    // ---- fold BN params ----
    bn_prep_kernel<<<1, 32,  0, stream>>>(bn1.g, bn1.b, bn1.m, bn1.v, bnp1, 5);
    for (int b = 0; b < 4; ++b) {
        bn_prep_kernel<<<1, 96,  0, stream>>>(BP[b].ba.g, BP[b].ba.b, BP[b].ba.m, BP[b].ba.v, bnpA[b], 96);
        bn_prep_kernel<<<1, 128, 0, stream>>>(BP[b].bb.g, BP[b].bb.b, BP[b].bb.m, BP[b].bb.v, bnpB[b], 128);
        bn_prep_kernel<<<1, 96,  0, stream>>>(BP[b].bo.g, BP[b].bo.b, BP[b].bo.m, BP[b].bo.v, bnpO[b], 96);
    }
    for (int i = 0; i < 4; ++i)
        bn_prep_kernel<<<1, 128, 0, stream>>>(PGs[i].bn.g, PGs[i].bn.b, PGs[i].bn.m, PGs[i].bn.v, bnpP[i], 128);

    // ---- pre-pack GEMM weights into fragment-ordered f16 arena ----
    size_t ho = 0;
    auto pack = [&](const float* src, int K, int Nout) -> const _Float16* {
        const int nK = (K + 31) >> 5, nT = (Nout + 15) >> 4;
        const int total = nT * nK * 512;
        _Float16* dst = arena + ho;
        ho += (size_t)((total + 127) & ~127);
        pack_w_kernel<<<(total + 255) / 256, 256, 0, stream>>>(src, dst, K, Nout);
        return dst;
    };
    const _Float16 *loW[4], *faW[4], *fbW[4], *foW[4], *pgW[4];
    for (int b = 0; b < 4; ++b) {
        int cin = (b == 0) ? 64 : 96;
        loW[b] = pack(BP[b].lo.w, cin + 44, 96);
        faW[b] = pack(BP[b].fa.w, 96, 128);
        fbW[b] = pack(BP[b].fb.w, 128, 96);
        foW[b] = pack(BP[b].fo.w, 384, 96);
    }
    for (int i = 0; i < 4; ++i)
        pgW[i] = pack(PGs[i].fc.w, (i == 0) ? 384 : 128, 128);
    const _Float16* o1W = pack(o1.w, 128, 64);
    const _Float16* o2W = pack(o2.w, 64, 64);
    const _Float16* o3W = pack(o3.w, 64, 4);

    // ---- input stage: bn1 -> global_exchange -> linear(20->64) ----
    stats5_kernel<<<NEV, 256, 0, stream>>>(x_in, bnp1, st5);
    inp_kernel<<<NHITS / 256, 256, 0, stream>>>(x_in, bnp1, st5, inp.w, inp.b, cur);

    // ---- 4 GravNet blocks ----
    int cin = 64;
    for (int b = 0; b < 4; ++b) {
        lin_sh_kernel<<<NHITS / 256, 256, 0, stream>>>(cur, cin,
            BP[b].ls.w, BP[b].ls.b, BP[b].lh.w, BP[b].lh.b, s4, h22);
        knn_kernel<<<NEV * (NPE / 4), 128, 0, stream>>>(s4, idx, wgt);
        aggregate_kernel<<<NHITS / 256, 256, 0, stream>>>(h22, idx, wgt, agg);

        // lin_o( cat[x, agg] ) -> x96
        gemm_wmma_kernel<0, false, true, false, false, false>
            <<<GEMM_GRID, 256, 0, stream>>>(cur, cin, cin, agg, 44, nullptr,
                loW[b], BP[b].lo.b, cin + 44, 96, nullptr, x96, 96, 0, nullptr, 0);
        // tanh(fc_a(bn_a(x96))) -> t128a
        gemm_wmma_kernel<1, true, false, false, false, false>
            <<<GEMM_GRID, 256, 0, stream>>>(x96, 96, 96, nullptr, 0, bnpA[b],
                faW[b], BP[b].fa.b, 96, 128, nullptr, t128a, 128, 0, nullptr, 0);
        // tanh(fc_b(bn_b(t128a))) -> x96
        gemm_wmma_kernel<1, true, false, false, false, false>
            <<<GEMM_GRID, 256, 0, stream>>>(t128a, 128, 128, nullptr, 0, bnpB[b],
                fbW[b], BP[b].fb.b, 128, 96, nullptr, x96, 96, 0, nullptr, 0);
        // global exchange stats of x96
        stats96_kernel<<<NEV * 96, 256, 0, stream>>>(x96, st96);
        // bn_out(tanh(fc_out(cat[stats, x96]))) -> outs[:, b*96:] and cur
        gemm_wmma_kernel<1, false, true, true, true, true>
            <<<GEMM_GRID, 256, 0, stream>>>(st96, 288, 288, x96, 96, nullptr,
                foW[b], BP[b].fo.b, 384, 96, bnpO[b], outs, 384, b * 96, cur, 96);
        cin = 96;
    }

    // ---- post-GN: 4x  bn(relu(linear)) ----
    const float* pin = outs; int pk = 384;
    float* pout = t128a;
    for (int i = 0; i < 4; ++i) {
        gemm_wmma_kernel<2, false, false, false, true, false>
            <<<GEMM_GRID, 256, 0, stream>>>(pin, pk, pk, nullptr, 0, nullptr,
                pgW[i], PGs[i].fc.b, pk, 128, bnpP[i], pout, 128, 0, nullptr, 0);
        pin = pout; pk = 128;
        pout = (pout == t128a) ? t128b : t128a;
    }

    // ---- output MLP ----
    gemm_wmma_kernel<2, false, false, false, false, false>
        <<<GEMM_GRID, 256, 0, stream>>>(pin, 128, 128, nullptr, 0, nullptr,
            o1W, o1.b, 128, 64, nullptr, t64a, 64, 0, nullptr, 0);
    gemm_wmma_kernel<2, false, false, false, false, false>
        <<<GEMM_GRID, 256, 0, stream>>>(t64a, 64, 64, nullptr, 0, nullptr,
            o2W, o2.b, 64, 64, nullptr, t64b, 64, 0, nullptr, 0);
    gemm_wmma_kernel<0, false, false, false, false, false>
        <<<GEMM_GRID, 256, 0, stream>>>(t64b, 64, 64, nullptr, 0, nullptr,
            o3W, o3.b, 64, 4, nullptr, (float*)d_out, 4, 0, nullptr, 0);
}
